// MultiHeadSelfAttention_3539053052370
// MI455X (gfx1250) — matrix operand order repaired and checked by reading, not compiled, not run on hardware
//
#include <hip/hip_runtime.h>
#include <hip/hip_bf16.h>

typedef __attribute__((ext_vector_type(16))) _Float16 v16h;
typedef __attribute__((ext_vector_type(8)))  _Float16 v8h;
typedef __attribute__((ext_vector_type(8)))  float    v8f;

union AFrag { v16h v; v8h h[2]; _Float16 e[16]; };
union CFrag { v8f  v; float e[8]; };

#define RSQRT_F 0.04419417382415922f   /* 1/sqrt(512) */
#define LDN 2048

// ---- WMMA wrapper: D = A(16x32 f16) * B(32x16 f16) + C(f32) ----
__device__ inline v8f wmma_f16(v16h a, v16h b, v8f c) {
    return __builtin_amdgcn_wmma_f32_16x16x32_f16(
        /*neg_a=*/false, a, /*neg_b=*/false, b,
        /*c_mod=*/(short)0, c, /*reuse_a=*/false, /*reuse_b=*/false);
}

// A fragment (16x32, f16) from row-major [row][k] with leading dim ld.
// ISA layout: lane<16 -> row=lane, elems 0..7 = K k0..k0+7, 8..15 = K k0+16..k0+23
//             lane>=16 -> row=lane-16, elems = K k0+8..15 and k0+24..31
__device__ inline v16h load_frag_a(const _Float16* __restrict__ base, int row0, int k0, int ld) {
    int lane = threadIdx.x & 31;
    int row  = row0 + (lane & 15);
    int kk   = k0 + ((lane >> 4) << 3);
    const _Float16* p = base + row * ld + kk;
    AFrag f;
    f.h[0] = *(const v8h*)(p);
    f.h[1] = *(const v8h*)(p + 16);
    return f.v;
}

// B fragment (32x16, f16) from column-major storage bt[col][k] (ld = col stride).
// ISA layout: lanes 0-15 = cols, K = k0..k0+7, k0+16..k0+23; lanes 16-31 = cols, K = k0+8..15, k0+24..31
__device__ inline v16h load_frag_b(const _Float16* __restrict__ bt, int k0, int col0, int ld) {
    int lane = threadIdx.x & 31;
    int col  = col0 + (lane & 15);
    int kk   = k0 + ((lane >> 4) << 3);
    const _Float16* p = bt + col * ld + kk;
    AFrag f;
    f.h[0] = *(const v8h*)(p);
    f.h[1] = *(const v8h*)(p + 16);
    return f.v;
}

// ---------------- conversion kernels ----------------
__global__ void cvt_x(const float* __restrict__ x, _Float16* __restrict__ xh, int n) {
    int i = blockIdx.x * blockDim.x + threadIdx.x;
    if (i < n) xh[i] = (_Float16)x[i];
}

// Transpose+convert all four weights into wT[w][col][f] (f16). Wq pre-scaled by 1/sqrt(F).
__global__ void cvt_w(const float* __restrict__ wq, const float* __restrict__ wk,
                      const float* __restrict__ wv, const float* __restrict__ wo,
                      _Float16* __restrict__ wT) {
    int i = blockIdx.x * blockDim.x + threadIdx.x;   // 4 * 512 * 512
    int w   = i >> 18;
    int rem = i & 262143;
    int f   = rem >> 9;
    int c   = rem & 511;
    const float* src = (w == 0) ? wq : (w == 1) ? wk : (w == 2) ? wv : wo;
    float v = src[f * 512 + c];
    if (w == 0) v *= RSQRT_F;
    wT[(w << 18) + c * 512 + f] = (_Float16)v;
}

// ---------------- QKV projection ----------------
// One wave computes a 32x64 output tile (2 row frags x 4 col frags, 8 accumulators).
// grid (8192/32=256, 512/64=8, 3 matrices).
// Q,K stored [b][h][n][64]; V stored transposed [b][h][64][n].
__global__ __launch_bounds__(32) void qkv_gemm(
        const _Float16* __restrict__ xh, const _Float16* __restrict__ wT,
        const float* __restrict__ bq, const float* __restrict__ bk, const float* __restrict__ bv,
        _Float16* __restrict__ Qb, _Float16* __restrict__ Kb, _Float16* __restrict__ Vt) {
    int mt = blockIdx.x, ct = blockIdx.y, z = blockIdx.z;
    int row0 = mt << 5, col0 = ct << 6;
    int lane = threadIdx.x & 31;
    const _Float16* wt = wT + z * 262144;
    const float* bias = (z == 0) ? bq : (z == 1) ? bk : bv;

    CFrag acc[2][4];
#pragma unroll
    for (int ci = 0; ci < 4; ++ci) {
        float bval = bias[col0 + ci * 16 + (lane & 15)];
        if (z == 0) bval *= RSQRT_F;
#pragma unroll
        for (int ri = 0; ri < 2; ++ri)
#pragma unroll
            for (int r = 0; r < 8; ++r) acc[ri][ci].e[r] = bval;
    }

    for (int k0 = 0; k0 < 512; k0 += 32) {
        v16h a0 = load_frag_a(xh, row0,      k0, 512);
        v16h a1 = load_frag_a(xh, row0 + 16, k0, 512);
#pragma unroll
        for (int ci = 0; ci < 4; ++ci) {
            v16h b = load_frag_b(wt, k0, col0 + ci * 16, 512);
            acc[0][ci].v = wmma_f16(a0, b, acc[0][ci].v);
            acc[1][ci].v = wmma_f16(a1, b, acc[1][ci].v);
        }
    }

    int rbase = (lane < 16) ? 0 : 8;
#pragma unroll
    for (int ri = 0; ri < 2; ++ri) {
#pragma unroll
        for (int ci = 0; ci < 4; ++ci) {
            int col = col0 + ci * 16 + (lane & 15);
            int h = col >> 6, dd = col & 63;
#pragma unroll
            for (int r = 0; r < 8; ++r) {
                int row = row0 + ri * 16 + rbase + r;
                int b_ = row >> 11, n = row & 2047;
                _Float16 val = (_Float16)acc[ri][ci].e[r];
                if (z == 0)      Qb[(((b_ << 3) + h) * 2048 + n) * 64 + dd] = val;
                else if (z == 1) Kb[(((b_ << 3) + h) * 2048 + n) * 64 + dd] = val;
                else             Vt[(((b_ << 3) + h) * 64 + dd) * 2048 + n] = val;
            }
        }
    }
}

// ---------------- attention core ----------------
// One WG per (b, h, 16-query tile). 4 waves.
// Dynamic LDS: S[16][2048] f32 (128KB) | P[16][2048] f16 (64KB) | red[128] | rowmax[16] | rowinv[16]
__global__ __launch_bounds__(128) void attention(
        const _Float16* __restrict__ Q, const _Float16* __restrict__ K,
        const _Float16* __restrict__ Vt, _Float16* __restrict__ attn) {
    extern __shared__ char smem[];
    float*    S      = (float*)smem;
    _Float16* P      = (_Float16*)(smem + 16 * LDN * 4);
    float*    red    = (float*)(smem + 16 * LDN * 4 + 16 * LDN * 2);
    float*    rowmax = red + 128;
    float*    rowinv = rowmax + 16;

    int bid = blockIdx.x;
    int it = bid & 127, h = (bid >> 7) & 7, b = bid >> 10;
    int i0 = it << 4;
    int tid = threadIdx.x, lane = tid & 31, w = tid >> 5;
    const _Float16* Qh = Q  + ((b * 8 + h) * 2048 + i0) * 64;
    const _Float16* Kh = K  + (size_t)(b * 8 + h) * 2048 * 64;
    const _Float16* Vh = Vt + (size_t)(b * 8 + h) * 64 * 2048;

    int col = lane & 15;
    int rbase = (lane < 16) ? 0 : 8;

    // Phase 1: S = Q * K^T (scale folded into Q). Q fragments hoisted (loop-invariant).
    {
        v16h qa0 = load_frag_a(Qh, 0, 0, 64);
        v16h qa1 = load_frag_a(Qh, 0, 32, 64);
        for (int jt = w * 512; jt < (w + 1) * 512; jt += 16) {
            CFrag c;
#pragma unroll
            for (int r = 0; r < 8; ++r) c.e[r] = 0.f;
            c.v = wmma_f16(qa0, load_frag_b(Kh, 0,  jt, 64), c.v);
            c.v = wmma_f16(qa1, load_frag_b(Kh, 32, jt, 64), c.v);
#pragma unroll
            for (int r = 0; r < 8; ++r) S[(rbase + r) * LDN + jt + col] = c.e[r];
        }
    }
    __syncthreads();

    // Phase 2: row softmax stats; P = exp(S - max) in f16 (unnormalized)
    {
        int row = tid >> 3, sub = tid & 7;   // 8 threads per row
        float m = -1e30f;
        for (int j = sub; j < LDN; j += 8) m = fmaxf(m, S[row * LDN + j]);
        red[tid] = m;
        __syncthreads();
        if (sub == 0) {
            float mm = red[tid];
#pragma unroll
            for (int t = 1; t < 8; ++t) mm = fmaxf(mm, red[tid + t]);
            rowmax[row] = mm;
        }
        __syncthreads();
        float mm = rowmax[row];
        float s = 0.f;
        for (int j = sub; j < LDN; j += 8) {
            float p = __expf(S[row * LDN + j] - mm);
            P[row * LDN + j] = (_Float16)p;
            s += p;
        }
        red[tid] = s;
        __syncthreads();
        if (sub == 0) {
            float ss = 0.f;
#pragma unroll
            for (int t = 0; t < 8; ++t) ss += red[tid + t];
            rowinv[row] = 1.0f / ss;
        }
        __syncthreads();
    }

    // Phase 3: out = (P @ V) * rowinv. Wave w owns d-columns [w*16, w*16+16).
    {
        CFrag c;
#pragma unroll
        for (int r = 0; r < 8; ++r) c.e[r] = 0.f;
        int d0 = w << 4;
        for (int j0 = 0; j0 < LDN; j0 += 32) {
            v16h a  = load_frag_a(P, 0, j0, LDN);    // A straight from LDS
            v16h bm = load_frag_b(Vh, j0, d0, LDN);  // Vt[d][n]: col = d, K = j contiguous
            c.v = wmma_f16(a, bm, c.v);
        }
#pragma unroll
        for (int r = 0; r < 8; ++r) {
            int row = rbase + r;
            float val = c.e[r] * rowinv[row];
            attn[(size_t)(b * 2048 + i0 + row) * 512 + h * 64 + d0 + col] = (_Float16)val;
        }
    }
}

// ---------------- output projection ----------------
// Same 32x64 register blocking as qkv_gemm. grid (256, 8).
__global__ __launch_bounds__(32) void out_gemm(
        const _Float16* __restrict__ attn, const _Float16* __restrict__ woT,
        const float* __restrict__ bo, float* __restrict__ out) {
    int mt = blockIdx.x, ct = blockIdx.y;
    int row0 = mt << 5, col0 = ct << 6;
    int lane = threadIdx.x & 31;

    CFrag acc[2][4];
#pragma unroll
    for (int ci = 0; ci < 4; ++ci) {
        float bval = bo[col0 + ci * 16 + (lane & 15)];
#pragma unroll
        for (int ri = 0; ri < 2; ++ri)
#pragma unroll
            for (int r = 0; r < 8; ++r) acc[ri][ci].e[r] = bval;
    }

    for (int k0 = 0; k0 < 512; k0 += 32) {
        v16h a0 = load_frag_a(attn, row0,      k0, 512);
        v16h a1 = load_frag_a(attn, row0 + 16, k0, 512);
#pragma unroll
        for (int ci = 0; ci < 4; ++ci) {
            v16h b = load_frag_b(woT, k0, col0 + ci * 16, 512);
            acc[0][ci].v = wmma_f16(a0, b, acc[0][ci].v);
            acc[1][ci].v = wmma_f16(a1, b, acc[1][ci].v);
        }
    }

    int rbase = (lane < 16) ? 0 : 8;
#pragma unroll
    for (int ri = 0; ri < 2; ++ri)
#pragma unroll
        for (int ci = 0; ci < 4; ++ci) {
            int colg = col0 + ci * 16 + (lane & 15);
#pragma unroll
            for (int r = 0; r < 8; ++r)
                out[(size_t)(row0 + ri * 16 + rbase + r) * 512 + colg] = acc[ri][ci].e[r];
        }
}

extern "C" void kernel_launch(void* const* d_in, const int* in_sizes, int n_in,
                              void* d_out, int out_size, void* d_ws, size_t ws_size,
                              hipStream_t stream) {
    (void)in_sizes; (void)n_in; (void)out_size; (void)ws_size;
    const float* x  = (const float*)d_in[0];
    const float* Wq = (const float*)d_in[1];
    const float* bq = (const float*)d_in[2];
    const float* Wk = (const float*)d_in[3];
    const float* bk = (const float*)d_in[4];
    const float* Wv = (const float*)d_in[5];
    const float* bv = (const float*)d_in[6];
    const float* Wo = (const float*)d_in[7];
    const float* bo = (const float*)d_in[8];
    float* out = (float*)d_out;

    char* ws = (char*)d_ws;
    _Float16* xh   = (_Float16*)(ws);               //  8,388,608 B : x f16 [8192][512]
    _Float16* wT   = (_Float16*)(ws + 8388608);     //  2,097,152 B : 4 transposed f16 weights
    _Float16* Qb   = (_Float16*)(ws + 10485760);    //  8,388,608 B : Q [b][h][n][64]
    _Float16* Kb   = (_Float16*)(ws + 18874368);    //  8,388,608 B : K [b][h][n][64]
    _Float16* Vt   = (_Float16*)(ws + 27262976);    //  8,388,608 B : V^T [b][h][64][n]
    _Float16* attn = (_Float16*)(ws + 35651584);    //  8,388,608 B : attn f16 [b][n][512]

    cvt_x<<<dim3(4194304 / 256), 256, 0, stream>>>(x, xh, 4194304);
    cvt_w<<<dim3(1048576 / 256), 256, 0, stream>>>(Wq, Wk, Wv, Wo, wT);
    qkv_gemm<<<dim3(256, 8, 3), 32, 0, stream>>>(xh, wT, bq, bk, bv, Qb, Kb, Vt);

    const int smem_bytes = 16 * 2048 * 4 + 16 * 2048 * 2 + 128 * 4 + 32 * 4; // 197,248 B
    hipFuncSetAttribute((const void*)attention,
                        hipFuncAttributeMaxDynamicSharedMemorySize, smem_bytes);
    attention<<<dim3(4096), 128, smem_bytes, stream>>>(Qb, Kb, Vt, attn);

    out_gemm<<<dim3(256, 8), 32, 0, stream>>>(attn, wT + 3 * 262144, bo, out);
}
